// AFTFull_48455821034241
// MI455X (gfx1250) — matrix operand order repaired and checked by reading, not compiled, not run on hardware
//
#include <hip/hip_runtime.h>

typedef __attribute__((ext_vector_type(16))) __bf16 v16bf;
typedef __attribute__((ext_vector_type(8)))  float  v8f;

#define DIM_ 256
#define HID_ 128
#define B_   8
#define T_   2048

__device__ __forceinline__ __bf16 bf(float f) { return (__bf16)f; }

union Frag16 { uint4 q[2]; v16bf v; };

// A-operand fragment for V_WMMA_F32_16X16X32_BF16 from a row-major [16][K..]
// bf16 LDS tile. Per ISA 7.12.2: lanes 0-15 hold M=lane, K halves {0..7,16..23};
// lanes 16-31 hold M=lane-16, K halves {8..15,24..31}. Consecutive half-pairs are
// contiguous in a row-major row -> two ds_load_b128 per lane.
__device__ __forceinline__ v16bf load_a_frag(const __bf16* tile, int rowStrideHalves, int lane) {
  int row = lane & 15;
  int lo  = (lane < 16) ? 0 : 8;
  const __bf16* rp = tile + row * rowStrideHalves;
  Frag16 f;
  f.q[0] = *reinterpret_cast<const uint4*>(rp + lo);
  f.q[1] = *reinterpret_cast<const uint4*>(rp + lo + 16);
  return f.v;
}

__device__ __forceinline__ v8f wmma_bf16(v16bf a, v16bf b, v8f c) {
  return __builtin_amdgcn_wmma_f32_16x16x32_bf16(false, a, false, b, (short)0, c, false, false);
}

// Packed f32x2 -> bf16x2 store (single b32 write, lowers to v_cvt_pk_bf16_f32).
__device__ __forceinline__ void store_bf16x2(__bf16* dst, float lo, float hi) {
  union { __bf16 h[2]; unsigned u; } p;
  p.h[0] = (__bf16)lo;
  p.h[1] = (__bf16)hi;
  *reinterpret_cast<unsigned*>(dst) = p.u;
}

// ---------------- K0: weights -> bf16, transposed row-major [K][N] -------------
__global__ __launch_bounds__(256) void prep_weights(const float* __restrict__ Wq,
                                                    const float* __restrict__ Wk,
                                                    const float* __restrict__ Wv,
                                                    const float* __restrict__ Wp,
                                                    __bf16* __restrict__ WqkvT,
                                                    __bf16* __restrict__ WpT) {
  int i = blockIdx.x * 256 + threadIdx.x;
  if (i < 256 * 384) {                      // WqkvT[d][n], n: 0-127 q, 128-255 k, 256-383 v
    int d = i / 384, n = i % 384;
    float v;
    if (n < 128)      v = Wq[n * 256 + d];
    else if (n < 256) v = Wk[(n - 128) * 256 + d];
    else              v = Wv[(n - 256) * 256 + d];
    WqkvT[i] = bf(v);
  } else {
    int j = i - 256 * 384;
    if (j < 128 * 256) {                    // WpT[h][d] = Wp[d][h]
      int h = j / 256, d = j % 256;
      WpT[j] = bf(Wp[d * 128 + h]);
    }
  }
}

// ---------------- K1: QKV projection via WMMA, sigmoid(Q) epilogue -------------
__global__ __launch_bounds__(256) void qkv_kernel(const float* __restrict__ x,
                                                  const float* __restrict__ bqv,
                                                  const float* __restrict__ bkv,
                                                  const float* __restrict__ bvv,
                                                  const __bf16* __restrict__ WqkvT,
                                                  float* __restrict__ Qsig,
                                                  float* __restrict__ Kbuf,
                                                  float* __restrict__ Vbuf) {
  __shared__ __align__(16) __bf16 xTile[16 * 40];
  int tid = threadIdx.x, lane = tid & 31, wv = tid >> 5;
  int bx = blockIdx.x, b = bx >> 7, t0 = (bx & 127) << 4;
  const float* xb = x + ((size_t)(b * T_ + t0)) * DIM_;

  v8f aq = {}; v8f ak = {}; v8f av = {};
  for (int kk = 0; kk < 8; ++kk) {
    {  // 16x32 chunk = 256 f32 pairs; one float2 load + packed b32 store per thread
      int r = tid >> 4, c = (tid & 15) << 1;
      float2 x2 = *reinterpret_cast<const float2*>(xb + (size_t)r * DIM_ + kk * 32 + c);
      store_bf16x2(&xTile[r * 40 + c], x2.x, x2.y);
    }
    __syncthreads();
    v16bf af = load_a_frag(xTile, 40, lane);
    v16bf b0, b1, b2;
#pragma unroll
    for (int s = 0; s < 16; ++s) {
      int kx = ((lane >> 4) << 3) + (s < 8 ? s : s + 8);
      const __bf16* wrow = WqkvT + (size_t)(kk * 32 + kx) * 384 + wv * 16 + (lane & 15);
      b0[s] = wrow[0];          // Q cols
      b1[s] = wrow[128];        // K cols
      b2[s] = wrow[256];        // V cols
    }
    aq = wmma_bf16(af, b0, aq);
    ak = wmma_bf16(af, b1, ak);
    av = wmma_bf16(af, b2, av);
    __syncthreads();
  }
  int h  = wv * 16 + (lane & 15);
  int rb = (lane < 16) ? 0 : 8;           // C/D: VGPR j -> M=j (lanes<16) / M=j+8
  float bqs = bqv[h], bks = bkv[h], bvs = bvv[h];
  for (int j = 0; j < 8; ++j) {
    size_t o = ((size_t)(b * T_ + t0 + rb + j)) * HID_ + h;
    float q = aq[j] + bqs;
    Qsig[o] = 1.f / (1.f + __expf(-q));
    Kbuf[o] = ak[j] + bks;
    Vbuf[o] = av[j] + bvs;
  }
}

// ---------------- K2a/b: softmax(K, axis=time) statistics ----------------------
__global__ __launch_bounds__(128) void ksm_partial(const float* __restrict__ Kbuf,
                                                   float2* __restrict__ part) {
  int b = blockIdx.x >> 3, ch = blockIdx.x & 7, h = threadIdx.x;
  float m = -3.402823466e38f, s = 0.f;
  const float* kp = Kbuf + ((size_t)(b * T_ + ch * 256)) * HID_ + h;
  for (int t = 0; t < 256; ++t) {
    float v = kp[(size_t)t * HID_];
    if (v > m) { s = s * __expf(m - v) + 1.f; m = v; }
    else       { s += __expf(v - m); }
  }
  part[(size_t)blockIdx.x * HID_ + h] = make_float2(m, s);
}

__global__ __launch_bounds__(256) void ksm_combine(const float2* __restrict__ part,
                                                   float* __restrict__ colmax,
                                                   float* __restrict__ colZinv) {
  int g = blockIdx.x * 256 + threadIdx.x;
  if (g >= B_ * HID_) return;
  int b = g >> 7, h = g & 127;
  float m = -3.402823466e38f, s = 0.f;
  for (int c = 0; c < 8; ++c) {
    float2 p = part[(size_t)(b * 8 + c) * HID_ + h];
    float nm = fmaxf(m, p.x);
    s = s * __expf(m - nm) + p.y * __expf(p.x - nm);
    m = nm;
  }
  colmax[g] = m;
  colZinv[g] = 1.f / s;
}

// ---------------- K2c: P[s][0:128]=eK*V, P[s][128:256]=eK (bf16, row-major) ----
__global__ __launch_bounds__(256) void build_p(const float* __restrict__ Kbuf,
                                               const float* __restrict__ Vbuf,
                                               const float* __restrict__ colmax,
                                               const float* __restrict__ colZinv,
                                               __bf16* __restrict__ P) {
  size_t i = (size_t)blockIdx.x * 256 + threadIdx.x;
  if (i >= (size_t)B_ * T_ * HID_) return;
  int h = (int)(i & 127);
  size_t bt = i >> 7;                    // b*T + t
  int b = (int)(bt >> 11);
  float ksm = __expf(Kbuf[i] - colmax[b * HID_ + h]) * colZinv[b * HID_ + h];
  float ek  = __expf(ksm);
  __bf16* prow = P + bt * 256;
  prow[h]       = bf(ek * Vbuf[i]);
  prow[128 + h] = bf(ek);
}

// Stage one 32x32 eab chunk (exp(softmax_row(adapt))) into a padded bf16 tile.
// 1024 elems = 512 f32 pairs; 256 threads x 2 pairs, fully unrolled, no bounds checks.
__device__ __forceinline__ void stage_eab(const float* __restrict__ arow, int k,
                                          __bf16* buf, const float* rowM,
                                          const float* rowZ, int tid) {
#pragma unroll
  for (int pp = 0; pp < 2; ++pp) {
    int e = tid + pp * 256;          // pair index
    int r = e >> 4;                  // 16 pairs per 32-wide row
    int c = (e & 15) << 1;
    float2 a2 = *reinterpret_cast<const float2*>(arow + (size_t)r * T_ + k * 32 + c);
    float m = rowM[r], zi = rowZ[r];
    float e0 = __expf(__expf(a2.x - m) * zi);   // eab = exp(softmax)
    float e1 = __expf(__expf(a2.y - m) * zi);
    store_bf16x2(&buf[r * 40 + c], e0, e1);
    if (k + 8 < 64) __builtin_prefetch(arow + (size_t)r * T_ + (k + 8) * 32 + c, 0, 0);
  }
}

// ---------------- K3: fused softmax(adapt)->exp GEMM -> Yt -> out projection ---
// M=32 block tile: each of 8 waves runs 4 WMMAs per K-step (2 row-frags x 2 col-frags).
__global__ __launch_bounds__(256) void aft_core(const float* __restrict__ adapt,
                                                const float* __restrict__ Qsig,
                                                const __bf16* __restrict__ P,
                                                const __bf16* __restrict__ WpT,
                                                const float* __restrict__ bp,
                                                float* __restrict__ out) {
  __shared__ float  rowM[32];
  __shared__ float  rowZ[32];
  __shared__ float2 red[256];
  __shared__ __align__(16) __bf16 aT[2][32 * 40];     // double-buffered A tile
  __shared__ float  ndTile[32 * 264];
  __shared__ __align__(16) __bf16 ytTile[32 * 136];

  int tid = threadIdx.x, lane = tid & 31, wv = tid >> 5;
  int bx = blockIdx.x, b = bx >> 6, t0 = (bx & 63) << 5;
  const float* arow = adapt + ((size_t)b * T_ + t0) * T_;

  // Pass 1: per-row online max / sum(exp); 8 threads per row, ~1 exp per element.
  {
    int r = tid >> 3, c0 = tid & 7;
    const float* ap = arow + (size_t)r * T_ + c0;
    float m = -3.402823466e38f, s = 0.f;
    for (int i = 0; i < 256; ++i) {
      float a = ap[i * 8];
      if (a > m) { s = s * __expf(m - a) + 1.f; m = a; }
      else       { s += __expf(a - m); }
    }
    red[tid] = make_float2(m, s);
    __syncthreads();
    if (c0 == 0) {
      float M = -3.402823466e38f, S = 0.f;
      for (int j = 0; j < 8; ++j) {
        float2 p = red[r * 8 + j];
        float nm = fmaxf(M, p.x);
        S = S * __expf(M - nm) + p.y * __expf(p.x - nm);
        M = nm;
      }
      rowM[r] = M;
      rowZ[r] = 1.f / S;
    }
    __syncthreads();
  }

  // Pass 2: 64 K-steps, double-buffered A staging, one barrier per step.
  v8f acc00 = {}; v8f acc01 = {}; v8f acc10 = {}; v8f acc11 = {};
  const __bf16* Pb = P + (size_t)b * T_ * 256;
  stage_eab(arow, 0, aT[0], rowM, rowZ, tid);
  for (int k = 0; k < 64; ++k) {
    __syncthreads();
    const __bf16* cur = aT[k & 1];
    v16bf a0 = load_a_frag(cur, 40, lane);                 // rows 0-15
    v16bf a1 = load_a_frag(cur + 16 * 40, 40, lane);       // rows 16-31
    v16bf b0, b1;
#pragma unroll
    for (int s = 0; s < 16; ++s) {
      int kx = ((lane >> 4) << 3) + (s < 8 ? s : s + 8);
      const __bf16* prow = Pb + (size_t)(k * 32 + kx) * 256 + wv * 32 + (lane & 15);
      b0[s] = prow[0];
      b1[s] = prow[16];
    }
    if (k < 63) stage_eab(arow, k + 1, aT[(k + 1) & 1], rowM, rowZ, tid);
    acc00 = wmma_bf16(a0, b0, acc00);
    acc01 = wmma_bf16(a0, b1, acc01);
    acc10 = wmma_bf16(a1, b0, acc10);
    acc11 = wmma_bf16(a1, b1, acc11);
  }

  // Spill num|den tile (cols 0-127 num, 128-255 den) to LDS.
  {
    int rb = (lane < 16) ? 0 : 8;
    int cl = lane & 15;
    for (int j = 0; j < 8; ++j) {
      ndTile[(rb + j) * 264 + wv * 32 + cl]             = acc00[j];
      ndTile[(rb + j) * 264 + wv * 32 + 16 + cl]        = acc01[j];
      ndTile[(16 + rb + j) * 264 + wv * 32 + cl]        = acc10[j];
      ndTile[(16 + rb + j) * 264 + wv * 32 + 16 + cl]   = acc11[j];
    }
  }
  __syncthreads();

  // Yt = sigmoid(Q) * num / den -> bf16 LDS tile [32][128]
  {
    int r = tid >> 3, h0 = (tid & 7) * 16;
    const float* qp = Qsig + ((size_t)(b * T_ + t0 + r)) * HID_ + h0;
#pragma unroll
    for (int i = 0; i < 16; i += 2) {
      float n0 = ndTile[r * 264 + h0 + i],     d0 = ndTile[r * 264 + 128 + h0 + i];
      float n1 = ndTile[r * 264 + h0 + i + 1], d1 = ndTile[r * 264 + 128 + h0 + i + 1];
      store_bf16x2(&ytTile[r * 136 + h0 + i], qp[i] * n0 / d0, qp[i + 1] * n1 / d1);
    }
  }
  __syncthreads();

  // Output projection: [32x128] x [128x256] bf16 WMMA, + bp
  v8f p00 = {}; v8f p01 = {}; v8f p10 = {}; v8f p11 = {};
  for (int kk = 0; kk < 4; ++kk) {
    v16bf a0 = load_a_frag(ytTile + kk * 32, 136, lane);
    v16bf a1 = load_a_frag(ytTile + 16 * 136 + kk * 32, 136, lane);
    v16bf b0, b1;
#pragma unroll
    for (int s = 0; s < 16; ++s) {
      int kx = ((lane >> 4) << 3) + (s < 8 ? s : s + 8);
      const __bf16* wrow = WpT + (size_t)(kk * 32 + kx) * DIM_ + wv * 32 + (lane & 15);
      b0[s] = wrow[0];
      b1[s] = wrow[16];
    }
    p00 = wmma_bf16(a0, b0, p00);
    p01 = wmma_bf16(a0, b1, p01);
    p10 = wmma_bf16(a1, b0, p10);
    p11 = wmma_bf16(a1, b1, p11);
  }
  {
    int rb = (lane < 16) ? 0 : 8;
    int cl = lane & 15;
    int c0 = wv * 32 + cl;
    float bp0 = bp[c0], bp1 = bp[c0 + 16];
    for (int j = 0; j < 8; ++j) {
      size_t o0 = ((size_t)(b * T_ + t0 + rb + j)) * DIM_;
      size_t o1 = ((size_t)(b * T_ + t0 + 16 + rb + j)) * DIM_;
      out[o0 + c0]      = p00[j] + bp0;
      out[o0 + c0 + 16] = p01[j] + bp1;
      out[o1 + c0]      = p10[j] + bp0;
      out[o1 + c0 + 16] = p11[j] + bp1;
    }
  }
}

// -------------------------------- launcher -------------------------------------
extern "C" void kernel_launch(void* const* d_in, const int* in_sizes, int n_in,
                              void* d_out, int out_size, void* d_ws, size_t ws_size,
                              hipStream_t stream) {
  const float* x     = (const float*)d_in[0];
  const float* adapt = (const float*)d_in[1];
  const float* Wq    = (const float*)d_in[2];
  const float* bq    = (const float*)d_in[3];
  const float* Wk    = (const float*)d_in[4];
  const float* bk    = (const float*)d_in[5];
  const float* Wv    = (const float*)d_in[6];
  const float* bv    = (const float*)d_in[7];
  const float* Wp    = (const float*)d_in[8];
  const float* bp    = (const float*)d_in[9];
  float* out = (float*)d_out;

  char* ws = (char*)d_ws;
  float*  Qsig    = (float*) (ws + 0);          //  8 MB  [B,T,H] f32
  float*  Kbuf    = (float*) (ws + 8388608);    //  8 MB  [B,T,H] f32
  float*  Vbuf    = (float*) (ws + 16777216);   //  8 MB  [B,T,H] f32
  __bf16* P       = (__bf16*)(ws + 25165824);   //  8 MB  [B,T,256] bf16 (eK*V | eK)
  __bf16* WqkvT   = (__bf16*)(ws + 33554432);   //  192KB [256,384] bf16
  __bf16* WpT     = (__bf16*)(ws + 33751040);   //  64 KB [128,256] bf16
  float2* part    = (float2*)(ws + 33816576);   //  64 KB [B,8,H] (m,s)
  float*  colmax  = (float*) (ws + 33882112);   //  4 KB
  float*  colZinv = (float*) (ws + 33886208);   //  4 KB

  prep_weights<<<512, 256, 0, stream>>>(Wq, Wk, Wv, Wp, WqkvT, WpT);
  qkv_kernel  <<<1024, 256, 0, stream>>>(x, bq, bk, bv, WqkvT, Qsig, Kbuf, Vbuf);
  ksm_partial <<<64, 128, 0, stream>>>(Kbuf, part);
  ksm_combine <<<4, 256, 0, stream>>>(part, colmax, colZinv);
  build_p     <<<8192, 256, 0, stream>>>(Kbuf, Vbuf, colmax, colZinv, P);
  aft_core    <<<512, 256, 0, stream>>>(adapt, Qsig, P, WpT, bp, out);
}
